// trajectory2seq_68985764708703
// MI455X (gfx1250) — matrix operand order repaired and checked by reading, not compiled, not run on hardware
//
#include <hip/hip_runtime.h>
#include <stdint.h>
#include <stddef.h>

#define H_DIM 512
#define G_DIM 2048      // 4*H
#define B_SZ  512
#define S_LEN 128
#define T_LEN 64
#define V_SZ  29
#define SOS_TOK 27

typedef __attribute__((ext_vector_type(16))) __bf16 v16bf;
typedef __attribute__((ext_vector_type(8)))  float  v8f;

struct __align__(16) U4 { unsigned int a, b, c, d; };
union Frag { v16bf v; U4 q[2]; };

__device__ inline unsigned short f2bf(float f) {
  union { float f; unsigned u; } v; v.f = f;
  unsigned r = v.u + 0x7FFFu + ((v.u >> 16) & 1u);
  return (unsigned short)(r >> 16);
}
__device__ inline float bf2f(unsigned short h) {
  union { unsigned u; float f; } v; v.u = ((unsigned)h) << 16; return v.f;
}
__device__ inline float sigm(float x) { return 1.0f / (1.0f + __expf(-x)); }

// ---- WMMA fragment loads (ISA 7.12.2 layouts, 16-bit A 16x32 / B 32x16) ----
// A: lane holds M = m0 + lane%16; elems 0..7 -> K = k0 + (lane/16)*8 + e,
//    elems 8..15 -> +16.  Two contiguous 16B loads.
__device__ inline v16bf load_a_frag(const unsigned short* A, int lda, int m0, int k0, int lane) {
  const unsigned short* p = A + (size_t)(m0 + (lane & 15)) * lda + k0 + ((lane >> 4) << 3);
  Frag f;
  f.q[0] = *(const U4*)p;
  f.q[1] = *(const U4*)(p + 16);
  return f.v;
}
// B (= W row-major N x K): lane holds N = n0 + lane%16; elems 0..7 -> K = k0 + (lane/16)*8 + e, elems 8..15 -> +16.
__device__ inline v16bf load_b_frag(const unsigned short* W, int ldw, int n0, int k0, int lane) {
  const unsigned short* p = W + (size_t)(n0 + (lane & 15)) * ldw + k0 + ((lane >> 4) << 3);
  Frag f;
  f.q[0] = *(const U4*)p;
  f.q[1] = *(const U4*)(p + 16);
  return f.v;
}

// ---- Generic WMMA GEMM: C[M x N] = A1@W1^T (+ A2@W2^T) + bias + addend ----
// A*: bf16 (M x K), W*: bf16 (N x K) row-major. M % 32 == 0; K % 32 == 0;
// W padded to >= ceil(N/64)*64 rows. Stores masked to n < Nstore.
__global__ __launch_bounds__(256) void gemm_wmma(
    const unsigned short* __restrict__ A1, int lda1,
    const unsigned short* __restrict__ W1, int ldw1, int K1,
    const unsigned short* __restrict__ A2, int lda2,
    const unsigned short* __restrict__ W2, int ldw2, int K2,
    const float* __restrict__ bias,
    const float* __restrict__ addend, int ldadd,
    float* __restrict__ Cf, unsigned short* __restrict__ Cb, int ldc,
    int M, int N, int Nstore)
{
  const int wave = threadIdx.x >> 5;
  const int lane = threadIdx.x & 31;
  const int tilesM = M >> 5;
  const int tilesN = (N + 63) >> 6;
  const int wt = blockIdx.x * 8 + wave;
  if (wt >= tilesM * tilesN) return;          // wave-uniform: EXEC stays all-ones
  const int M0 = (wt % tilesM) << 5;
  const int N0 = (wt / tilesM) << 6;

  v8f acc[2][4];
  const v8f vzero = {0.f,0.f,0.f,0.f,0.f,0.f,0.f,0.f};
#pragma unroll
  for (int i = 0; i < 2; ++i)
#pragma unroll
    for (int j = 0; j < 4; ++j) acc[i][j] = vzero;

  for (int k0 = 0; k0 < K1; k0 += 32) {
    v16bf a0 = load_a_frag(A1, lda1, M0,      k0, lane);
    v16bf a1 = load_a_frag(A1, lda1, M0 + 16, k0, lane);
#pragma unroll
    for (int j = 0; j < 4; ++j) {
      v16bf b = load_b_frag(W1, ldw1, N0 + (j << 4), k0, lane);
      acc[0][j] = __builtin_amdgcn_wmma_f32_16x16x32_bf16(false, a0, false, b, (short)0, acc[0][j], false, false);
      acc[1][j] = __builtin_amdgcn_wmma_f32_16x16x32_bf16(false, a1, false, b, (short)0, acc[1][j], false, false);
    }
  }
  if (K2 > 0) {
    for (int k0 = 0; k0 < K2; k0 += 32) {
      v16bf a0 = load_a_frag(A2, lda2, M0,      k0, lane);
      v16bf a1 = load_a_frag(A2, lda2, M0 + 16, k0, lane);
#pragma unroll
      for (int j = 0; j < 4; ++j) {
        v16bf b = load_b_frag(W2, ldw2, N0 + (j << 4), k0, lane);
        acc[0][j] = __builtin_amdgcn_wmma_f32_16x16x32_bf16(false, a0, false, b, (short)0, acc[0][j], false, false);
        acc[1][j] = __builtin_amdgcn_wmma_f32_16x16x32_bf16(false, a1, false, b, (short)0, acc[1][j], false, false);
      }
    }
  }

  // C layout: VGPR e, lanes 0-15 -> row e, lanes 16-31 -> row e+8; col = lane%16
  const int mh = (lane & 16) >> 1;   // 0 or 8
  const int nl = lane & 15;
#pragma unroll
  for (int i = 0; i < 2; ++i) {
#pragma unroll
    for (int j = 0; j < 4; ++j) {
      const int n = N0 + (j << 4) + nl;
      if (n >= Nstore) continue;
      const float badd = bias ? bias[n] : 0.f;
#pragma unroll
      for (int e = 0; e < 8; ++e) {
        const int m = M0 + (i << 4) + mh + e;
        float v = acc[i][j][e] + badd;
        if (addend) v += addend[(size_t)m * ldadd + n];
        if (Cf) Cf[(size_t)m * ldc + n] = v;
        if (Cb) Cb[(size_t)m * ldc + n] = f2bf(v);
      }
    }
  }
}

// ---- elementwise / small kernels ----
__global__ void cvt_pad_bf16(const float* __restrict__ src, unsigned short* __restrict__ dst,
                             int rows, int cols, int rowsP, int colsP) {
  int idx = blockIdx.x * blockDim.x + threadIdx.x;
  if (idx >= rowsP * colsP) return;
  int r = idx / colsP, c = idx % colsP;
  float v = (r < rows && c < cols) ? src[(size_t)r * cols + c] : 0.f;
  dst[idx] = f2bf(v);
}

__global__ void pad_bias_f32(const float* __restrict__ src, float* __restrict__ dst, int n, int npad) {
  int i = blockIdx.x * blockDim.x + threadIdx.x;
  if (i < npad) dst[i] = (i < n) ? src[i] : 0.f;
}

// Xih[b][n] = x[b][t][0]*W[n][0] + x[b][t][1]*W[n][1]   (enc0 K=2 input proj, fp32)
__global__ void xih_kernel(const float* __restrict__ x, const float* __restrict__ Wih,
                           float* __restrict__ out, int t) {
  int idx = blockIdx.x * blockDim.x + threadIdx.x;
  if (idx >= B_SZ * G_DIM) return;
  int b = idx / G_DIM, n = idx % G_DIM;
  float x0 = x[((size_t)b * S_LEN + t) * 2 + 0];
  float x1 = x[((size_t)b * S_LEN + t) * 2 + 1];
  out[idx] = x0 * Wih[n * 2] + x1 * Wih[n * 2 + 1];
}

// LSTM cell elementwise: gates (B x 4H, order i,f,g,o); updates c, h(fp32+bf16)
__global__ void lstm_cell(const float* __restrict__ gates, float* __restrict__ c,
                          float* __restrict__ hf, unsigned short* __restrict__ hb) {
  int idx = blockIdx.x * blockDim.x + threadIdx.x;
  if (idx >= B_SZ * H_DIM) return;
  int b = idx / H_DIM, j = idx % H_DIM;
  const float* g = gates + (size_t)b * G_DIM;
  float ig = sigm(g[j]);
  float fg = sigm(g[H_DIM + j]);
  float gg = tanhf(g[2 * H_DIM + j]);
  float og = sigm(g[3 * H_DIM + j]);
  float cn = fg * c[idx] + ig * gg;
  c[idx] = cn;
  float h = og * tanhf(cn);
  hf[idx] = h;
  hb[idx] = f2bf(h);
}

// scores[b*S+s] = dot(q[b,:], values[b,s,:]) — one wave per (b,s)
__global__ void attn_scores(const float* __restrict__ q, const unsigned short* __restrict__ values,
                            float* __restrict__ scores) {
  int wave = threadIdx.x >> 5, lane = threadIdx.x & 31;
  int bs = blockIdx.x * (blockDim.x >> 5) + wave;
  if (bs >= B_SZ * S_LEN) return;
  int b = bs / S_LEN;
  const unsigned short* vp = values + (size_t)bs * H_DIM;
  const float* qp = q + (size_t)b * H_DIM;
  float acc = 0.f;
  for (int h = lane; h < H_DIM; h += 32) acc += qp[h] * bf2f(vp[h]);
#pragma unroll
  for (int off = 16; off; off >>= 1) acc += __shfl_xor(acc, off, 32);
  if (lane == 0) scores[bs] = acc;
}

__global__ void softmax128(const float* __restrict__ scores, float* __restrict__ aw) {
  __shared__ float sm[S_LEN];
  int b = blockIdx.x, s = threadIdx.x;
  float v = scores[(size_t)b * S_LEN + s];
  sm[s] = v; __syncthreads();
  for (int off = 64; off; off >>= 1) { if (s < off) sm[s] = fmaxf(sm[s], sm[s + off]); __syncthreads(); }
  float mx = sm[0]; __syncthreads();
  float e = __expf(v - mx);
  sm[s] = e; __syncthreads();
  for (int off = 64; off; off >>= 1) { if (s < off) sm[s] += sm[s + off]; __syncthreads(); }
  aw[(size_t)b * S_LEN + s] = e / sm[0];
}

// ao[b][h] = sum_s aw[b][s] * values[b][s][h]
__global__ void attn_out(const float* __restrict__ aw, const unsigned short* __restrict__ values,
                         float* __restrict__ aof, unsigned short* __restrict__ aob) {
  int idx = blockIdx.x * blockDim.x + threadIdx.x;
  if (idx >= B_SZ * H_DIM) return;
  int b = idx / H_DIM, h = idx % H_DIM;
  const float* a = aw + (size_t)b * S_LEN;
  const unsigned short* vp = values + (size_t)b * S_LEN * H_DIM + h;
  float acc = 0.f;
  for (int s = 0; s < S_LEN; ++s) acc += a[s] * bf2f(vp[(size_t)s * H_DIM]);
  aof[idx] = acc;
  aob[idx] = f2bf(acc);
}

// next token: tf ? target[b][t] : argmax_h(out[b,:])  (first-max tie-break)
__global__ void argmax_next(const float* __restrict__ hf, const int* __restrict__ target,
                            int* __restrict__ tok, int t, int tf) {
  __shared__ float sv[256];
  __shared__ int   si[256];
  int b = blockIdx.x, tid = threadIdx.x;
  float v0 = hf[(size_t)b * H_DIM + tid];
  float v1 = hf[(size_t)b * H_DIM + 256 + tid];
  float v; int idx;
  if (v1 > v0) { v = v1; idx = tid + 256; } else { v = v0; idx = tid; }
  sv[tid] = v; si[tid] = idx; __syncthreads();
  for (int off = 128; off; off >>= 1) {
    if (tid < off) {
      if (sv[tid + off] > sv[tid] || (sv[tid + off] == sv[tid] && si[tid + off] < si[tid])) {
        sv[tid] = sv[tid + off]; si[tid] = si[tid + off];
      }
    }
    __syncthreads();
  }
  if (tid == 0) {
    int am = si[0]; if (am > V_SZ - 1) am = V_SZ - 1;   // clip (jnp.take mode='clip')
    tok[b] = tf ? target[(size_t)b * T_LEN + t] : am;
  }
}

__global__ void embed_kernel(const unsigned short* __restrict__ emb, const int* __restrict__ tok,
                             unsigned short* __restrict__ xe) {
  int idx = blockIdx.x * blockDim.x + threadIdx.x;
  if (idx >= B_SZ * H_DIM) return;
  int b = idx / H_DIM, h = idx % H_DIM;
  int tk = tok[b];
  if (tk < 0) tk = 0; if (tk > V_SZ - 1) tk = V_SZ - 1;
  xe[idx] = emb[(size_t)tk * H_DIM + h];
}

__global__ void init_tok(int* __restrict__ tok) {
  int i = blockIdx.x * blockDim.x + threadIdx.x;
  if (i < B_SZ) tok[i] = SOS_TOK;
}

// ---- host-side threefry2x32 (exact JAX semantics, key = (0, 42)) ----
static void threefry2x32(uint32_t k0, uint32_t k1, uint32_t c0, uint32_t c1,
                         uint32_t* o0, uint32_t* o1) {
  const unsigned R[8] = {13, 15, 26, 6, 17, 29, 16, 24};
  uint32_t ks[3] = {k0, k1, k0 ^ k1 ^ 0x1BD11BDAu};
  uint32_t x0 = c0 + ks[0], x1 = c1 + ks[1];
  for (int g = 0; g < 5; ++g) {
    for (int r = 0; r < 4; ++r) {
      unsigned rot = R[(g & 1) * 4 + r];
      x0 += x1; x1 = (x1 << rot) | (x1 >> (32 - rot)); x1 ^= x0;
    }
    x0 += ks[(g + 1) % 3];
    x1 += ks[(g + 2) % 3] + (uint32_t)(g + 1);
  }
  *o0 = x0; *o1 = x1;
}

extern "C" void kernel_launch(void* const* d_in, const int* in_sizes, int n_in,
                              void* d_out, int out_size, void* d_ws, size_t ws_size,
                              hipStream_t stream) {
  (void)in_sizes; (void)n_in; (void)out_size;
  const float* x      = (const float*)d_in[0];
  const int*   target = (const int*)d_in[1];
  const float* e0_Wih = (const float*)d_in[2];
  const float* e0_Whh = (const float*)d_in[3];
  const float* e0_b   = (const float*)d_in[4];
  const float* e1_Wih = (const float*)d_in[5];
  const float* e1_Whh = (const float*)d_in[6];
  const float* e1_b   = (const float*)d_in[7];
  const float* dc0_Wih = (const float*)d_in[8];
  const float* dc0_Whh = (const float*)d_in[9];
  const float* dc0_b   = (const float*)d_in[10];
  const float* dc1_Wih = (const float*)d_in[11];
  const float* dc1_Whh = (const float*)d_in[12];
  const float* dc1_b   = (const float*)d_in[13];
  const float* emb  = (const float*)d_in[14];
  const float* Wq   = (const float*)d_in[15];
  const float* bq   = (const float*)d_in[16];
  const float* Wv   = (const float*)d_in[17];
  const float* bv   = (const float*)d_in[18];
  const float* Wfc  = (const float*)d_in[19];
  const float* bfc  = (const float*)d_in[20];
  const float* Wfc1 = (const float*)d_in[21];
  const float* bfc1 = (const float*)d_in[22];
  float* out = (float*)d_out;

  // workspace bump allocator (256B aligned)
  char* base = (char*)d_ws;
  size_t off = 0;
  auto alloc = [&](size_t bytes) -> void* {
    void* p = base + off;
    off = (off + bytes + 255) & ~(size_t)255;
    return p;
  };
  unsigned short* whh0b  = (unsigned short*)alloc((size_t)G_DIM * H_DIM * 2);
  unsigned short* wih1b  = (unsigned short*)alloc((size_t)G_DIM * H_DIM * 2);
  unsigned short* whh1b  = (unsigned short*)alloc((size_t)G_DIM * H_DIM * 2);
  unsigned short* wihd0b = (unsigned short*)alloc((size_t)G_DIM * H_DIM * 2);
  unsigned short* whhd0b = (unsigned short*)alloc((size_t)G_DIM * H_DIM * 2);
  unsigned short* wihd1b = (unsigned short*)alloc((size_t)G_DIM * H_DIM * 2);
  unsigned short* whhd1b = (unsigned short*)alloc((size_t)G_DIM * H_DIM * 2);
  unsigned short* wqb    = (unsigned short*)alloc((size_t)H_DIM * H_DIM * 2);
  unsigned short* wvb    = (unsigned short*)alloc((size_t)H_DIM * 32 * 2);       // 512 x 32 (K pad 29->32)
  unsigned short* wfcb   = (unsigned short*)alloc((size_t)64 * H_DIM * 2);       // rows pad 29->64
  unsigned short* wfc1b  = (unsigned short*)alloc((size_t)64 * 2 * H_DIM * 2);   // rows pad 29->64, K=1024
  unsigned short* embb   = (unsigned short*)alloc((size_t)V_SZ * H_DIM * 2);
  float* bfcp  = (float*)alloc(64 * 4);
  float* bfc1p = (float*)alloc(64 * 4);
  float* h0f = (float*)alloc((size_t)B_SZ * H_DIM * 4);
  float* h1f = (float*)alloc((size_t)B_SZ * H_DIM * 4);
  float* c0  = (float*)alloc((size_t)B_SZ * H_DIM * 4);
  float* c1  = (float*)alloc((size_t)B_SZ * H_DIM * 4);
  unsigned short* h0b = (unsigned short*)alloc((size_t)B_SZ * H_DIM * 2);
  unsigned short* h1b = (unsigned short*)alloc((size_t)B_SZ * H_DIM * 2);
  float* gates = (float*)alloc((size_t)B_SZ * G_DIM * 4);
  float* xih   = (float*)alloc((size_t)B_SZ * G_DIM * 4);
  unsigned short* enc_out = (unsigned short*)alloc((size_t)B_SZ * S_LEN * 32 * 2); // (B*S) x 32, pad zeroed
  unsigned short* values  = (unsigned short*)alloc((size_t)B_SZ * S_LEN * H_DIM * 2);
  float* qb     = (float*)alloc((size_t)B_SZ * H_DIM * 4);
  float* scores = (float*)alloc((size_t)B_SZ * S_LEN * 4);
  float* aw     = (float*)alloc((size_t)B_SZ * S_LEN * 4);
  float* aof    = (float*)alloc((size_t)B_SZ * H_DIM * 4);
  unsigned short* aob = (unsigned short*)alloc((size_t)B_SZ * H_DIM * 2);
  unsigned short* xeb = (unsigned short*)alloc((size_t)B_SZ * H_DIM * 2);
  int* tok = (int*)alloc((size_t)B_SZ * 4);
  if (off > ws_size) return;  // workspace too small; deterministic no-op

  // teacher-forcing mask: jax.random.uniform(key(42),(64,)) < 0.6 (threefry2x32)
  uint32_t bits[T_LEN];
  for (int i = 0; i < 32; ++i) {
    uint32_t o0, o1;
    threefry2x32(0u, 42u, (uint32_t)i, (uint32_t)(i + 32), &o0, &o1);
    bits[i] = o0; bits[i + 32] = o1;
  }
  int tf[T_LEN];
  for (int t = 0; t < T_LEN; ++t) {
    union { uint32_t u; float f; } v; v.u = (bits[t] >> 9) | 0x3F800000u;
    tf[t] = ((v.f - 1.0f) < 0.6f) ? 1 : 0;
  }

  auto cvt = [&](const float* src, unsigned short* dst, int r, int c, int rp, int cp) {
    int tot = rp * cp;
    cvt_pad_bf16<<<(tot + 255) / 256, 256, 0, stream>>>(src, dst, r, c, rp, cp);
  };
  auto gemm = [&](const unsigned short* A1, int lda1, const unsigned short* W1, int ldw1, int K1,
                  const unsigned short* A2, int lda2, const unsigned short* W2, int ldw2, int K2,
                  const float* bias, const float* addend, int ldadd,
                  float* Cf, unsigned short* Cb, int ldc, int M, int N, int Nstore) {
    int tilesM = M >> 5, tilesN = (N + 63) >> 6;
    int blocks = (tilesM * tilesN + 7) >> 3;
    gemm_wmma<<<blocks, 256, 0, stream>>>(A1, lda1, W1, ldw1, K1, A2, lda2, W2, ldw2, K2,
                                          bias, addend, ldadd, Cf, Cb, ldc, M, N, Nstore);
  };

  // ---- weight conversion to bf16 (zero-padded), bias padding ----
  cvt(e0_Whh,  whh0b,  G_DIM, H_DIM, G_DIM, H_DIM);
  cvt(e1_Wih,  wih1b,  G_DIM, H_DIM, G_DIM, H_DIM);
  cvt(e1_Whh,  whh1b,  G_DIM, H_DIM, G_DIM, H_DIM);
  cvt(dc0_Wih, wihd0b, G_DIM, H_DIM, G_DIM, H_DIM);
  cvt(dc0_Whh, whhd0b, G_DIM, H_DIM, G_DIM, H_DIM);
  cvt(dc1_Wih, wihd1b, G_DIM, H_DIM, G_DIM, H_DIM);
  cvt(dc1_Whh, whhd1b, G_DIM, H_DIM, G_DIM, H_DIM);
  cvt(Wq,   wqb,   H_DIM, H_DIM, H_DIM, H_DIM);
  cvt(Wv,   wvb,   H_DIM, V_SZ,  H_DIM, 32);
  cvt(Wfc,  wfcb,  V_SZ,  H_DIM, 64,    H_DIM);
  cvt(Wfc1, wfc1b, V_SZ,  2 * H_DIM, 64, 2 * H_DIM);
  cvt(emb,  embb,  V_SZ,  H_DIM, V_SZ,  H_DIM);
  pad_bias_f32<<<1, 64, 0, stream>>>(bfc,  bfcp,  V_SZ, 64);
  pad_bias_f32<<<1, 64, 0, stream>>>(bfc1, bfc1p, V_SZ, 64);

  // ---- state init ----
  hipMemsetAsync(h0f, 0, (size_t)B_SZ * H_DIM * 4, stream);
  hipMemsetAsync(h1f, 0, (size_t)B_SZ * H_DIM * 4, stream);
  hipMemsetAsync(c0,  0, (size_t)B_SZ * H_DIM * 4, stream);
  hipMemsetAsync(c1,  0, (size_t)B_SZ * H_DIM * 4, stream);
  hipMemsetAsync(h0b, 0, (size_t)B_SZ * H_DIM * 2, stream);
  hipMemsetAsync(h1b, 0, (size_t)B_SZ * H_DIM * 2, stream);

  const int EW = (B_SZ * H_DIM + 255) / 256;   // elementwise grid over B*H

  // ---- encoder scan ----
  for (int t = 0; t < S_LEN; ++t) {
    xih_kernel<<<(B_SZ * G_DIM + 255) / 256, 256, 0, stream>>>(x, e0_Wih, xih, t);
    // gates0 = h0 @ Whh0^T + Xih_t + b0
    gemm(h0b, H_DIM, whh0b, H_DIM, H_DIM, nullptr, 0, nullptr, 0, 0,
         e0_b, xih, G_DIM, gates, nullptr, G_DIM, B_SZ, G_DIM, G_DIM);
    lstm_cell<<<EW, 256, 0, stream>>>(gates, c0, h0f, h0b);
    // gates1 = h0_new @ Wih1^T + h1 @ Whh1^T + b1
    gemm(h0b, H_DIM, wih1b, H_DIM, H_DIM, h1b, H_DIM, whh1b, H_DIM, H_DIM,
         e1_b, nullptr, 0, gates, nullptr, G_DIM, B_SZ, G_DIM, G_DIM);
    lstm_cell<<<EW, 256, 0, stream>>>(gates, c1, h1f, h1b);
    // enc_out_t = h1 @ Wfc^T + bfc  -> bf16, row (b*S+t), stride 32 (cols 29..31 = 0)
    gemm(h1b, H_DIM, wfcb, H_DIM, H_DIM, nullptr, 0, nullptr, 0, 0,
         bfcp, nullptr, 0, nullptr, enc_out + (size_t)t * 32, S_LEN * 32, B_SZ, 64, 32);
  }

  // hn snapshot -> d_out (before decoder mutates states)
  const size_t OUT_HN = (size_t)B_SZ * T_LEN * V_SZ;          // 950272
  const size_t OUT_ATTN = OUT_HN + (size_t)2 * B_SZ * H_DIM;  // 1474560
  hipMemcpyAsync(out + OUT_HN, h0f, (size_t)B_SZ * H_DIM * 4, hipMemcpyDeviceToDevice, stream);
  hipMemcpyAsync(out + OUT_HN + (size_t)B_SZ * H_DIM, h1f, (size_t)B_SZ * H_DIM * 4,
                 hipMemcpyDeviceToDevice, stream);

  // values = enc_out @ Wv^T + bv   (M = B*S, N = H, K = 32 padded) -> bf16
  gemm(enc_out, 32, wvb, 32, 32, nullptr, 0, nullptr, 0, 0,
       bv, nullptr, 0, nullptr, values, H_DIM, B_SZ * S_LEN, H_DIM, H_DIM);

  // ---- decoder scan ----
  init_tok<<<2, 256, 0, stream>>>(tok);
  for (int t = 0; t < T_LEN; ++t) {
    embed_kernel<<<EW, 256, 0, stream>>>(embb, tok, xeb);
    gemm(xeb, H_DIM, wihd0b, H_DIM, H_DIM, h0b, H_DIM, whhd0b, H_DIM, H_DIM,
         dc0_b, nullptr, 0, gates, nullptr, G_DIM, B_SZ, G_DIM, G_DIM);
    lstm_cell<<<EW, 256, 0, stream>>>(gates, c0, h0f, h0b);
    gemm(h0b, H_DIM, wihd1b, H_DIM, H_DIM, h1b, H_DIM, whhd1b, H_DIM, H_DIM,
         dc1_b, nullptr, 0, gates, nullptr, G_DIM, B_SZ, G_DIM, G_DIM);
    lstm_cell<<<EW, 256, 0, stream>>>(gates, c1, h1f, h1b);
    // q = out @ Wq^T + bq
    gemm(h1b, H_DIM, wqb, H_DIM, H_DIM, nullptr, 0, nullptr, 0, 0,
         bq, nullptr, 0, qb, nullptr, H_DIM, B_SZ, H_DIM, H_DIM);
    attn_scores<<<(B_SZ * S_LEN + 7) / 8, 256, 0, stream>>>(qb, values, scores);
    softmax128<<<B_SZ, S_LEN, 0, stream>>>(scores, aw);
    attn_out<<<EW, 256, 0, stream>>>(aw, values, aof, aob);
    // logits = [out, ao] @ Wfc1^T + bfc1 -> d_out vec_out[b][t][:]
    gemm(h1b, H_DIM, wfc1b, 2 * H_DIM, H_DIM, aob, H_DIM, wfc1b + H_DIM, 2 * H_DIM, H_DIM,
         bfc1p, nullptr, 0, out + (size_t)t * V_SZ, nullptr, T_LEN * V_SZ, B_SZ, 64, V_SZ);
    argmax_next<<<B_SZ, 256, 0, stream>>>(h1f, target, tok, t, tf[t]);
  }

  // attn_last = last-step attention weights
  hipMemcpyAsync(out + OUT_ATTN, aw, (size_t)B_SZ * S_LEN * 4, hipMemcpyDeviceToDevice, stream);
}
